// SAGEModel_60155311948300
// MI455X (gfx1250) — matrix operand order repaired and checked by reading, not compiled, not run on hardware
//
#include <hip/hip_runtime.h>
#include <cstdint>

// ---------------------------------------------------------------------------
// GraphSAGE (3-layer, D=128) for MI455X / gfx1250, wave32.
//   - node features kept in bf16 (halves the dominant edge-gather traffic)
//   - edge scatter uses CDNA5 global_atomic_pk_add_bf16 (2 ch / atomic)
//   - 1/deg folded into the scatter (rdeg[dst] constant per destination)
//   - weights pre-swizzled once into WMMA B-fragment order, staged to LDS
//     per block via global_load_async_to_lds_b128 (ASYNCcnt path)
//   - two v_wmma_f32_16x16x32_bf16 per k-step (agg@Wl + h@Wr), f32 accum
// ---------------------------------------------------------------------------

#define N_NODES   100000
#define N_EDGES   800000
#define D         128
#define N_LAYERS  3
#define N_CLASSES 10
#define N_GRAPHS  64
#define BN_EPS    1e-5f

typedef __attribute__((ext_vector_type(16))) __bf16 v16bf;
typedef __attribute__((ext_vector_type(8)))  __bf16 v8bf;
typedef __attribute__((ext_vector_type(8)))  float  v8f;

// f32 -> bf16, round-to-nearest-even
__device__ __forceinline__ __bf16 cvt_bf16(float f) {
  unsigned u = __builtin_bit_cast(unsigned, f);
  u += 0x7FFFu + ((u >> 16) & 1u);
  unsigned short h = (unsigned short)(u >> 16);
  return __builtin_bit_cast(__bf16, h);
}
__device__ __forceinline__ unsigned pack2_bf16(float a, float b) {
  unsigned ua = __builtin_bit_cast(unsigned, a);
  ua += 0x7FFFu + ((ua >> 16) & 1u);
  unsigned ub = __builtin_bit_cast(unsigned, b);
  ub += 0x7FFFu + ((ub >> 16) & 1u);
  return (ua >> 16) | (ub & 0xFFFF0000u);
}
__device__ __forceinline__ float bfu_lo(unsigned u) {
  return __builtin_bit_cast(float, u << 16);
}
__device__ __forceinline__ float bfu_hi(unsigned u) {
  return __builtin_bit_cast(float, u & 0xFFFF0000u);
}

// ---------------------------------------------------------------- utilities
__global__ void fill0_kernel(float* __restrict__ p, int n) {
  int i = blockIdx.x * blockDim.x + threadIdx.x;
  int stride = gridDim.x * blockDim.x;
  for (; i < n; i += stride) p[i] = 0.0f;
}

__global__ void degree_kernel(const long long* __restrict__ dst,
                              float* __restrict__ deg, int n_edges) {
  int e = blockIdx.x * blockDim.x + threadIdx.x;
  if (e < n_edges) atomicAdd(&deg[(int)dst[e]], 1.0f);
}

__global__ void recip_deg_kernel(float* __restrict__ deg, int n) {
  int i = blockIdx.x * blockDim.x + threadIdx.x;
  if (i < n) deg[i] = 1.0f / fmaxf(deg[i], 1.0f);
}

// f32 stream -> packed bf16 pairs
__global__ void f32_to_bf16_kernel(const float* __restrict__ in,
                                   unsigned* __restrict__ out, int n_pairs) {
  int i = blockIdx.x * blockDim.x + threadIdx.x;
  if (i >= n_pairs) return;
  float2 v = ((const float2*)in)[i];
  out[i] = pack2_bf16(v.x, v.y);
}

// ------------------------------------------------- weight fragment swizzle
// B fragment layout (32x16 bf16, per cdna5 CDNA5 ISA sparse-B pattern):
//   fragment f = ((t*4 + kb)*32 + lane), 16 bf16 (32B) per lane:
//   N = t*16 + lane%16, K = kb*32 + (e<8?0:16) + (lane>=16?8:0) + (e&7).
// 6 matrices: {Wl0,Wr0,Wl1,Wr1,Wl2,Wr2}, each 1024 fragments (32KB bf16).
__global__ void swizzle_w_kernel(const float* __restrict__ Wl,
                                 const float* __restrict__ Wr,
                                 __bf16* __restrict__ wfrag) {
  int m     = blockIdx.x >> 2;            // matrix 0..5
  int layer = m >> 1;
  const float* src = ((m & 1) ? Wr : Wl) + (size_t)layer * D * D;
  int f     = (blockIdx.x & 3) * 256 + threadIdx.x;   // fragment 0..1023
  int t     = f >> 7;
  int rem   = f & 127;
  int kb    = rem >> 5;
  int lane  = rem & 31;
  int n     = t * 16 + (lane & 15);
  int kbase = kb * 32 + (lane >> 4) * 8;
  __bf16* dstp = wfrag + (size_t)m * (D * D) + (size_t)f * 16;
  #pragma unroll
  for (int e = 0; e < 16; ++e)
    dstp[e] = cvt_bf16(src[(kbase + (e & 7) + (e >> 3) * 16) * D + n]);
}

// ----------------------------------------------------------- edge aggregate
// One wave per edge; lane moves 4 bf16 channels (8B gather) and scatters with
// two packed-bf16 atomics, pre-scaled by rdeg[dst] (mean aggregation).
__global__ __launch_bounds__(256)
void aggregate_kernel(const __bf16* __restrict__ h,
                      const long long* __restrict__ src,
                      const long long* __restrict__ dst,
                      const float* __restrict__ rdeg,
                      __bf16* __restrict__ agg, int n_edges) {
  long long t = (long long)blockIdx.x * blockDim.x + threadIdx.x;
  int e = (int)(t >> 5);
  if (e >= n_edges) return;
  int lane = threadIdx.x & 31;
  int s = (int)src[e];
  int d = (int)dst[e];
  float rd = rdeg[d];
  uint2 v = ((const uint2*)(h + (size_t)s * D))[lane];
  unsigned p0 = pack2_bf16(bfu_lo(v.x) * rd, bfu_hi(v.x) * rd);
  unsigned p1 = pack2_bf16(bfu_lo(v.y) * rd, bfu_hi(v.y) * rd);
  unsigned long long a0 =
      (unsigned long long)(agg + (size_t)d * D + lane * 4);
  asm volatile("global_atomic_pk_add_bf16 %0, %1, off scope:SCOPE_DEV"
               :: "v"(a0), "v"(p0) : "memory");
  asm volatile("global_atomic_pk_add_bf16 %0, %1, off offset:4 scope:SCOPE_DEV"
               :: "v"(a0), "v"(p1) : "memory");
}

// ----------------------------------------------------------------- SAGE GEMM
// out[m,n] = sum_k agg[m,k]*Wl[k,n] + h[m,k]*Wr[k,n] + bl[n]
// Block: 256 threads = 8 waves; block -> 16-row M tile; wave w -> N tile w.
// A fragment (16x32 bf16): lane row M=lane%16,
//   element e -> K = (e<8?0:16) + (lane>=16?8:0) + (e&7)  (two 16B runs).
// B fragments are pre-swizzled; one aligned 32B LDS read per fragment.
__global__ __launch_bounds__(256)
void sage_gemm_kernel(const __bf16* __restrict__ agg,
                      const __bf16* __restrict__ h,
                      const __bf16* __restrict__ wfL,
                      const __bf16* __restrict__ wfR,
                      const float* __restrict__ bl,
                      float* __restrict__ out) {
  __shared__ __attribute__((aligned(32))) __bf16 sW[2 * D * D];  // 64 KB

  // Stage both fragment-ordered weight matrices (32KB each) into LDS with
  // CDNA5 async global->LDS copies (tracked by ASYNCcnt).
  {
    unsigned lds_base = (unsigned)(unsigned long long)(size_t)&sW[0];
    const char* gl = (const char*)wfL;
    const char* gr = (const char*)wfR;
    #pragma unroll
    for (int it = 0; it < 8; ++it) {               // 8 x 4KB per matrix
      unsigned o = threadIdx.x * 16u + (unsigned)it * 4096u;
      unsigned long long ga = (unsigned long long)(gl + o);
      unsigned ld0 = lds_base + o;
      asm volatile("global_load_async_to_lds_b128 %0, %1, off"
                   :: "v"(ld0), "v"(ga) : "memory");
      unsigned long long gb = (unsigned long long)(gr + o);
      unsigned ld1 = lds_base + 32768u + o;
      asm volatile("global_load_async_to_lds_b128 %0, %1, off"
                   :: "v"(ld1), "v"(gb) : "memory");
    }
    asm volatile("s_wait_asynccnt 0" ::: "memory");
  }
  __syncthreads();

  const int m0   = blockIdx.x * 16;
  const int wave = threadIdx.x >> 5;
  const int n0   = wave * 16;
  const int lane = threadIdx.x & 31;
  const int half = lane >> 4;
  const int lm   = lane & 15;
  const int rowA = m0 + lm;

  const v16bf* fragsL = (const v16bf*)sW;          // 1024 fragments
  const v16bf* fragsR = (const v16bf*)(sW + D * D);

  v8f acc = {};

  #pragma unroll
  for (int kb = 0; kb < 4; ++kb) {
    const int k0 = kb * 32;
    const __bf16* pa = agg + (size_t)rowA * D + k0 + half * 8;
    const __bf16* ph = h   + (size_t)rowA * D + k0 + half * 8;
    v8bf a_lo = *(const v8bf*)pa;
    v8bf a_hi = *(const v8bf*)(pa + 16);
    v8bf h_lo = *(const v8bf*)ph;
    v8bf h_hi = *(const v8bf*)(ph + 16);
    v16bf fa = __builtin_shufflevector(a_lo, a_hi,
        0, 1, 2, 3, 4, 5, 6, 7, 8, 9, 10, 11, 12, 13, 14, 15);
    v16bf fh = __builtin_shufflevector(h_lo, h_hi,
        0, 1, 2, 3, 4, 5, 6, 7, 8, 9, 10, 11, 12, 13, 14, 15);

    v16bf fbl = fragsL[(wave * 4 + kb) * 32 + lane];
    v16bf fbr = fragsR[(wave * 4 + kb) * 32 + lane];

    acc = __builtin_amdgcn_wmma_f32_16x16x32_bf16(
        false, fa, false, fbl, (short)0, acc, false, false);
    acc = __builtin_amdgcn_wmma_f32_16x16x32_bf16(
        false, fh, false, fbr, (short)0, acc, false, false);
  }

  const float bias = bl[n0 + lm];
  #pragma unroll
  for (int r = 0; r < 8; ++r)
    out[(size_t)(m0 + r + half * 8) * D + n0 + lm] = acc[r] + bias;
}

// ------------------------------------------------------------ batchnorm+relu
__global__ void stats_kernel(const float* __restrict__ h,
                             float* __restrict__ sums, int n_rows) {
  int c = threadIdx.x;                 // 128 threads
  float s = 0.0f, sq = 0.0f;
  for (int r = blockIdx.x; r < n_rows; r += gridDim.x) {
    float v = h[(size_t)r * D + c];
    s += v; sq += v * v;
  }
  atomicAdd(&sums[c],     s);
  atomicAdd(&sums[D + c], sq);
}

// normalize + ReLU + repack to bf16 pairs for the next layer
__global__ void bn_relu_pack_kernel(const float* __restrict__ hg,
                                    const float* __restrict__ sums,
                                    const float* __restrict__ gamma,
                                    const float* __restrict__ beta,
                                    unsigned* __restrict__ hout, int n_pairs) {
  int i = blockIdx.x * blockDim.x + threadIdx.x;
  if (i >= n_pairs) return;
  int c0 = (i & 63) * 2;
  const float inv_n = 1.0f / (float)N_NODES;
  float2 v = ((const float2*)hg)[i];
  float mu0 = sums[c0] * inv_n;
  float mu1 = sums[c0 + 1] * inv_n;
  float va0 = sums[D + c0] * inv_n - mu0 * mu0;
  float va1 = sums[D + c0 + 1] * inv_n - mu1 * mu1;
  float r0 = fmaxf((v.x - mu0) * rsqrtf(va0 + BN_EPS) * gamma[c0] + beta[c0], 0.0f);
  float r1 = fmaxf((v.y - mu1) * rsqrtf(va1 + BN_EPS) * gamma[c0 + 1] + beta[c0 + 1], 0.0f);
  hout[i] = pack2_bf16(r0, r1);
}

// ------------------------------------------------------------------ pooling
__global__ __launch_bounds__(256)
void pool_accum_kernel(const __bf16* __restrict__ h,
                       const long long* __restrict__ batch,
                       float* __restrict__ pool,
                       float* __restrict__ cnt, int n_nodes) {
  long long t = (long long)blockIdx.x * blockDim.x + threadIdx.x;
  int v = (int)(t >> 5);
  if (v >= n_nodes) return;
  int lane = threadIdx.x & 31;
  int g = (int)batch[v];
  uint2 val = ((const uint2*)(h + (size_t)v * D))[lane];
  float* pp = pool + (size_t)g * D + lane * 4;
  atomicAdd(pp + 0, bfu_lo(val.x));
  atomicAdd(pp + 1, bfu_hi(val.x));
  atomicAdd(pp + 2, bfu_lo(val.y));
  atomicAdd(pp + 3, bfu_hi(val.y));
  if (lane == 0) atomicAdd(&cnt[g], 1.0f);
}

__global__ void classify_kernel(const float* __restrict__ pool,
                                const float* __restrict__ cnt,
                                const float* __restrict__ Wc,
                                const float* __restrict__ bc,
                                float* __restrict__ out) {
  int idx = blockIdx.x * blockDim.x + threadIdx.x;
  if (idx >= N_GRAPHS * N_CLASSES) return;
  int g = idx / N_CLASSES, k = idx % N_CLASSES;
  float inv = 1.0f / fmaxf(cnt[g], 1.0f);
  float s = bc[k];
  #pragma unroll 8
  for (int c = 0; c < D; ++c)
    s += pool[(size_t)g * D + c] * inv * Wc[c * N_CLASSES + k];
  out[idx] = s;
}

// ---------------------------------------------------------------------------
extern "C" void kernel_launch(void* const* d_in, const int* in_sizes, int n_in,
                              void* d_out, int out_size, void* d_ws, size_t ws_size,
                              hipStream_t stream) {
  (void)in_sizes; (void)n_in; (void)out_size; (void)ws_size;

  const float*     x     = (const float*)d_in[0];
  const long long* eidx  = (const long long*)d_in[1];   // int64 (2, N_EDGES)
  const long long* batch = (const long long*)d_in[2];   // int64 (N_NODES)
  const float*     Wl    = (const float*)d_in[3];       // (3,128,128)
  const float*     bl    = (const float*)d_in[4];       // (3,128)
  const float*     Wr    = (const float*)d_in[5];       // (3,128,128)
  const float*     gamma = (const float*)d_in[6];       // (3,128)
  const float*     beta  = (const float*)d_in[7];       // (3,128)
  const float*     Wc    = (const float*)d_in[8];       // (128,10)
  const float*     bc    = (const float*)d_in[9];       // (10,)
  float*           out   = (float*)d_out;               // (64,10)

  const long long* src = eidx;
  const long long* dst = eidx + N_EDGES;

  // Workspace layout (f32 units, all 16B-aligned offsets).
  float*  ws    = (float*)d_ws;
  float*  rdeg  = ws;                               // 100,000  (pad 100,032)
  __bf16* wfrag = (__bf16*)(ws + 100032);           // 6*16384 bf16 = 49,152 f32
  __bf16* agg   = (__bf16*)(ws + 149184);           // 12.8M bf16 = 6.4M f32
  __bf16* hA    = (__bf16*)(ws + 6549184);          // 6.4M f32
  __bf16* hB    = (__bf16*)(ws + 12949184);         // 6.4M f32
  float*  hG    = ws + 19349184;                    // 12.8M f32
  float*  stats = ws + 32149184;                    // 256
  float*  pool  = ws + 32149440;                    // 8192
  float*  cnt   = ws + 32157632;                    // 64

  const int feat_pairs = N_NODES * (D / 2);         // 6.4M packed bf16 pairs

  // Degrees -> reciprocal degrees (in place).
  fill0_kernel<<<256, 256, 0, stream>>>(rdeg, N_NODES);
  degree_kernel<<<(N_EDGES + 255) / 256, 256, 0, stream>>>(dst, rdeg, N_EDGES);
  recip_deg_kernel<<<(N_NODES + 255) / 256, 256, 0, stream>>>(rdeg, N_NODES);

  // x -> bf16 input features; weights -> fragment order (once).
  f32_to_bf16_kernel<<<(feat_pairs + 255) / 256, 256, 0, stream>>>(
      x, (unsigned*)hA, feat_pairs);
  swizzle_w_kernel<<<24, 256, 0, stream>>>(Wl, Wr, wfrag);

  const __bf16* h_in  = hA;
  __bf16*       h_out = hB;
  for (int l = 0; l < N_LAYERS; ++l) {
    fill0_kernel<<<2048, 256, 0, stream>>>((float*)agg, feat_pairs);
    aggregate_kernel<<<(N_EDGES * 32) / 256, 256, 0, stream>>>(
        h_in, src, dst, rdeg, agg, N_EDGES);
    sage_gemm_kernel<<<N_NODES / 16, 256, 0, stream>>>(
        agg, h_in, wfrag + (size_t)(2 * l) * (D * D),
        wfrag + (size_t)(2 * l + 1) * (D * D), bl + l * D, hG);
    fill0_kernel<<<1, 256, 0, stream>>>(stats, 2 * D);
    stats_kernel<<<256, D, 0, stream>>>(hG, stats, N_NODES);
    bn_relu_pack_kernel<<<(feat_pairs + 255) / 256, 256, 0, stream>>>(
        hG, stats, gamma + l * D, beta + l * D, (unsigned*)h_out, feat_pairs);
    const __bf16* nin = h_out;
    h_out = (l == 0) ? hA : ((l == 1) ? hB : h_out);
    // simple ping-pong: layer0 hA->hB, layer1 hB->hA, layer2 hA->hB
    h_in = nin;
    h_out = (h_in == hA) ? hB : hA;
  }
  // h_in now points at the final bf16 node embeddings.

  fill0_kernel<<<64, 256, 0, stream>>>(pool, N_GRAPHS * D + N_GRAPHS);
  pool_accum_kernel<<<(N_NODES * 32 + 255) / 256, 256, 0, stream>>>(
      h_in, batch, pool, cnt, N_NODES);
  classify_kernel<<<(N_GRAPHS * N_CLASSES + 255) / 256, 256, 0, stream>>>(
      pool, cnt, Wc, bc, out);
}
